// DeepPolyLeakyReLU_40183714021526
// MI455X (gfx1250) — matrix operand order repaired and checked by reading, not compiled, not run on hardware
//
#include <hip/hip_runtime.h>
#include <hip/hip_bf16.h>
#include <cstdint>
#include <cstddef>

// DeepPoly 2-layer certifier, MI455X (gfx1250, wave32, WMMA).
// M = W2 @ W1 (4096^3) via split-bf16 V_WMMA_F32_16X16X32_BF16 (hi*hi + hi*lo
// + lo*hi, f32 accumulate), fused with interval reduction
// lb[j] = sum_i M[j,i]*mid[i] - |M[j,i]|*rad[i]  (ub with +).
// Fast path: bf16 hi/lo planes precomputed (W1 transposed) + double-buffered
// GLOBAL_LOAD_ASYNC_TO_LDS_B128 pipeline (ASYNCcnt). Fallback splits in-loop.

#define NI 4096
#define NS 0.01f
#define BM 128
#define BN 128
#define BK 32
#define LDA 40   // padded LDS row stride (bf16 elems): 80B rows, 16B aligned, conflict-free
#define KCHUNKS (NI / BN)

typedef __attribute__((ext_vector_type(16))) __bf16 v16bf;
typedef __attribute__((ext_vector_type(8)))  float  v8f;

struct __align__(16) U4 { unsigned x, y, z, w; };
struct __align__(8)  U2 { unsigned x, y; };

union Frag {
  U4    q[2];
  v16bf v;
};

__device__ __forceinline__ unsigned short bf16_rne(float x) {
  unsigned u = __float_as_uint(x);
  u += 0x7fffu + ((u >> 16) & 1u);
  return (unsigned short)(u >> 16);
}
__device__ __forceinline__ float bf16_to_f(unsigned short h) {
  return __uint_as_float(((unsigned)h) << 16);
}
__device__ __forceinline__ void split_bf16(float a, unsigned short &hi, unsigned short &lo) {
  hi = bf16_rne(a);
  float r = a - bf16_to_f(hi);
  lo = bf16_rne(r);
}

// Async global->LDS copy, 16B per lane, tracked by ASYNCcnt (GVS addressing:
// 64-bit SGPR base + 32-bit per-lane byte offset; vdst VGPR = LDS byte address).
__device__ __forceinline__ void async_b128(unsigned lds, unsigned goff,
                                           const unsigned short *base) {
  asm volatile("global_load_async_to_lds_b128 %0, %1, %2"
               :: "v"(lds), "v"(goff), "s"(base)
               : "memory");
}

// ---------------------------------------------------------------------------
// Kernel 1: zero-fill the whole output (two NxN diag matrices + 2 intercept rows)
// ---------------------------------------------------------------------------
__global__ void fill_zero_kernel(float4 *__restrict__ out, long n4) {
  long i = (long)blockIdx.x * blockDim.x + threadIdx.x;
  const long stride = (long)gridDim.x * blockDim.x;
  const float4 z = make_float4(0.f, 0.f, 0.f, 0.f);
  for (; i < n4; i += stride) out[i] = z;
}

// ---------------------------------------------------------------------------
// Kernel 2: c[j] = sum_p b1[p] * W2[j,p] + b2[j]   (one wave per row)
// ---------------------------------------------------------------------------
__global__ void __launch_bounds__(256) bias_fold_kernel(
    const float *__restrict__ W2, const float *__restrict__ b1,
    const float *__restrict__ b2, float *__restrict__ wsC) {
  const int lane = threadIdx.x & 31;
  const int wid  = threadIdx.x >> 5;
  const int j    = blockIdx.x * 8 + wid;
  const float *row = W2 + (size_t)j * NI;
  float s = 0.f;
  for (int p = lane; p < NI; p += 32) s += row[p] * b1[p];
  #pragma unroll
  for (int m = 16; m >= 1; m >>= 1) s += __shfl_xor(s, m, 32);
  if (lane == 0) wsC[j] = s + b2[j];
}

// ---------------------------------------------------------------------------
// Kernel 3a: split W2 -> bf16 hi/lo planes (row-major [j][p])
// ---------------------------------------------------------------------------
__global__ void __launch_bounds__(256) split_w2_kernel(
    const float *__restrict__ W2,
    unsigned short *__restrict__ Ah, unsigned short *__restrict__ Al) {
  const size_t i = ((size_t)blockIdx.x * 256 + threadIdx.x) * 4;
  const float4 v = *(const float4 *)(W2 + i);
  unsigned short h0, l0, h1, l1, h2, l2, h3, l3;
  split_bf16(v.x, h0, l0); split_bf16(v.y, h1, l1);
  split_bf16(v.z, h2, l2); split_bf16(v.w, h3, l3);
  U2 ph = { (unsigned)h0 | ((unsigned)h1 << 16), (unsigned)h2 | ((unsigned)h3 << 16) };
  U2 pl = { (unsigned)l0 | ((unsigned)l1 << 16), (unsigned)l2 | ((unsigned)l3 << 16) };
  *(U2 *)(Ah + i) = ph;
  *(U2 *)(Al + i) = pl;
}

// ---------------------------------------------------------------------------
// Kernel 3b: split + transpose W1 -> bf16 hi/lo planes in [n][k] layout
// ---------------------------------------------------------------------------
__global__ void __launch_bounds__(256) split_w1t_kernel(
    const float *__restrict__ W1,
    unsigned short *__restrict__ BTh, unsigned short *__restrict__ BTl) {
  __shared__ float tile[64][65];
  const int x0 = blockIdx.x * 64;        // cols of W1 (n)
  const int y0 = blockIdx.y * 64;        // rows of W1 (k)
  const int tx = threadIdx.x & 63;
  const int ty = threadIdx.x >> 6;       // 0..3
  #pragma unroll
  for (int i = 0; i < 16; ++i) {
    const int r = ty + 4 * i;
    tile[r][tx] = W1[(size_t)(y0 + r) * NI + x0 + tx];
  }
  __syncthreads();
  #pragma unroll
  for (int i = 0; i < 16; ++i) {
    const int n = ty + 4 * i;
    const float v = tile[tx][n];
    unsigned short h, l;
    split_bf16(v, h, l);
    const size_t o = (size_t)(x0 + n) * NI + y0 + tx;   // [n][k], k contiguous
    BTh[o] = h;
    BTl[o] = l;
  }
}

// ---------------------------------------------------------------------------
// Kernel 4a (fast path): async double-buffered WMMA GEMM + fused reduction.
// ---------------------------------------------------------------------------
__global__ void __launch_bounds__(256) gemm_bounds_async_kernel(
    const unsigned short *__restrict__ Ah, const unsigned short *__restrict__ Al,
    const unsigned short *__restrict__ BTh, const unsigned short *__restrict__ BTl,
    const float *__restrict__ lb0, const float *__restrict__ ub0,
    float *__restrict__ wsLB, float *__restrict__ wsUB) {
  __shared__ unsigned short sT[2][4][BM * LDA];   // [buf][plane: Ah,Al,Bh,Bl]
  __shared__ float sLB[BM];
  __shared__ float sUB[BM];

  const int t  = threadIdx.x;
  const int jt = blockIdx.x;
  const int it = blockIdx.y;
  const int j0 = jt * BM;
  const int i0 = it * BN;

  if (t < BM) sLB[t] = 0.f; else sUB[t - BM] = 0.f;

  const int lane    = t & 31;
  const int wid     = t >> 5;
  const int waveM   = wid & 3;
  const int waveN   = wid >> 2;
  const int m0      = waveM * 32;
  const int n0      = waveN * 64;
  const int lrow    = lane & 15;
  const int halfsel = lane >> 4;

  // staging map: row = t/2, k offset = (t%2)*16 (8 bf16 = one b128 per plane)
  const int pr_row = t >> 1;
  const int pr_ko  = (t & 1) * 16;
  const int so     = pr_row * LDA + pr_ko;

  unsigned ldsAddr[2][4];
  #pragma unroll
  for (int b = 0; b < 2; ++b)
    #pragma unroll
    for (int pl = 0; pl < 4; ++pl)
      ldsAddr[b][pl] = (unsigned)(uintptr_t)&sT[b][pl][so];

  const unsigned gA = (unsigned)(((j0 + pr_row) * NI + pr_ko) * 2);  // byte offsets
  const unsigned gB = (unsigned)(((i0 + pr_row) * NI + pr_ko) * 2);

  v8f acc[2][4] = {};

  auto prefetch = [&](int buf, unsigned ob) {
    async_b128(ldsAddr[buf][0], gA + ob, Ah);
    async_b128(ldsAddr[buf][1], gA + ob, Al);
    async_b128(ldsAddr[buf][2], gB + ob, BTh);
    async_b128(ldsAddr[buf][3], gB + ob, BTl);
  };

  auto compute_step = [&](int buf) {
    const int a_kbase = halfsel * 8;
    Frag fah[2], fal[2];
    #pragma unroll
    for (int mi = 0; mi < 2; ++mi) {
      const int r = (m0 + mi * 16 + lrow) * LDA;
      fah[mi].q[0] = *(const U4 *)&sT[buf][0][r + a_kbase];
      fah[mi].q[1] = *(const U4 *)&sT[buf][0][r + a_kbase + 16];
      fal[mi].q[0] = *(const U4 *)&sT[buf][1][r + a_kbase];
      fal[mi].q[1] = *(const U4 *)&sT[buf][1][r + a_kbase + 16];
    }
    const int b_kbase = halfsel * 8;
    #pragma unroll
    for (int ni = 0; ni < 4; ++ni) {
      const int c = (n0 + ni * 16 + lrow) * LDA;
      Frag fbh, fbl;
      fbh.q[0] = *(const U4 *)&sT[buf][2][c + b_kbase];
      fbh.q[1] = *(const U4 *)&sT[buf][2][c + b_kbase + 16];
      fbl.q[0] = *(const U4 *)&sT[buf][3][c + b_kbase];
      fbl.q[1] = *(const U4 *)&sT[buf][3][c + b_kbase + 16];
      #pragma unroll
      for (int mi = 0; mi < 2; ++mi) {
        acc[mi][ni] = __builtin_amdgcn_wmma_f32_16x16x32_bf16(
            false, fah[mi].v, false, fbh.v, (short)0, acc[mi][ni], false, false);
        acc[mi][ni] = __builtin_amdgcn_wmma_f32_16x16x32_bf16(
            false, fah[mi].v, false, fbl.v, (short)0, acc[mi][ni], false, false);
        acc[mi][ni] = __builtin_amdgcn_wmma_f32_16x16x32_bf16(
            false, fal[mi].v, false, fbh.v, (short)0, acc[mi][ni], false, false);
      }
    }
  };

  constexpr int NSTEP = NI / BK;        // 128 (even)
  constexpr unsigned STEPB = BK * 2;    // 64 bytes per k-step

  prefetch(0, 0);
  for (int i = 0; i < NSTEP; i += 2) {
    const unsigned o1 = (unsigned)(i + 1) * STEPB;               // always in range
    prefetch(1, o1);
    asm volatile("s_wait_asynccnt 0x4" ::: "memory");            // buf0 resident
    __syncthreads();
    compute_step(0);
    __syncthreads();
    const unsigned o2 = (unsigned)((i + 2) % NSTEP) * STEPB;     // wraps to dummy 0 at tail
    prefetch(0, o2);
    asm volatile("s_wait_asynccnt 0x4" ::: "memory");            // buf1 resident
    __syncthreads();
    compute_step(1);
    __syncthreads();
  }
  asm volatile("s_wait_asynccnt 0x0" ::: "memory");              // drain tail dummies

  // ---- fused epilogue: per-row partial lb/ub over this 128-col chunk ----
  float midv[4], radv[4];
  #pragma unroll
  for (int ni = 0; ni < 4; ++ni) {
    const int c = i0 + n0 + ni * 16 + lrow;
    const float lo = lb0[c], hi = ub0[c];
    midv[ni] = 0.5f * (lo + hi);
    radv[ni] = 0.5f * (hi - lo);
  }
  #pragma unroll
  for (int mi = 0; mi < 2; ++mi) {
    #pragma unroll
    for (int v = 0; v < 8; ++v) {
      float lbp = 0.f, ubp = 0.f;
      #pragma unroll
      for (int ni = 0; ni < 4; ++ni) {
        const float m  = acc[mi][ni][v];
        const float t1 = m * midv[ni];
        const float t2 = fabsf(m) * radv[ni];
        lbp += t1 - t2;
        ubp += t1 + t2;
      }
      #pragma unroll
      for (int s = 8; s >= 1; s >>= 1) {
        lbp += __shfl_xor(lbp, s, 32);
        ubp += __shfl_xor(ubp, s, 32);
      }
      if (lrow == 0) {
        const int r = m0 + mi * 16 + v + halfsel * 8;
        atomicAdd(&sLB[r], lbp);   // ds_add_f32
        atomicAdd(&sUB[r], ubp);
      }
    }
  }
  __syncthreads();
  if (t < BM) wsLB[(size_t)it * NI + j0 + t] = sLB[t];
  else        wsUB[(size_t)it * NI + j0 + (t - BM)] = sUB[t - BM];
}

// ---------------------------------------------------------------------------
// Kernel 4b (fallback, small ws): self-contained, splits f32 in-loop.
// ---------------------------------------------------------------------------
__global__ void __launch_bounds__(256) gemm_bounds_fused_kernel(
    const float *__restrict__ W2, const float *__restrict__ W1,
    const float *__restrict__ lb0, const float *__restrict__ ub0,
    float *__restrict__ wsLB, float *__restrict__ wsUB) {
  __shared__ unsigned short sAh[BM * LDA];
  __shared__ unsigned short sAl[BM * LDA];
  __shared__ unsigned short sBh[BN * LDA];
  __shared__ unsigned short sBl[BN * LDA];
  __shared__ float sLB[BM];
  __shared__ float sUB[BM];

  const int t  = threadIdx.x;
  const int jt = blockIdx.x;
  const int it = blockIdx.y;
  const int j0 = jt * BM;
  const int i0 = it * BN;

  if (t < BM) sLB[t] = 0.f; else sUB[t - BM] = 0.f;

  const int lane    = t & 31;
  const int wid     = t >> 5;
  const int waveM   = wid & 3;
  const int waveN   = wid >> 2;
  const int m0      = waveM * 32;
  const int n0      = waveN * 64;
  const int lrow    = lane & 15;
  const int halfsel = lane >> 4;

  const int a_row = t >> 3;
  const int a_c4  = t & 7;
  const int b_col = t & 127;
  const int b_kh  = t >> 7;

  v8f acc[2][4] = {};

  for (int p0 = 0; p0 < NI; p0 += BK) {
    float4 aReg[4];
    #pragma unroll
    for (int r = 0; r < 4; ++r)
      aReg[r] = *(const float4 *)(W2 + (size_t)(j0 + r * 32 + a_row) * NI + p0 + a_c4 * 4);
    float bReg[16];
    #pragma unroll
    for (int kk = 0; kk < 16; ++kk)
      bReg[kk] = W1[(size_t)(p0 + b_kh * 16 + kk) * NI + i0 + b_col];

    __syncthreads();

    #pragma unroll
    for (int r = 0; r < 4; ++r) {
      unsigned short h0, l0, h1, l1, h2, l2, h3, l3;
      split_bf16(aReg[r].x, h0, l0); split_bf16(aReg[r].y, h1, l1);
      split_bf16(aReg[r].z, h2, l2); split_bf16(aReg[r].w, h3, l3);
      U2 ph = { (unsigned)h0 | ((unsigned)h1 << 16), (unsigned)h2 | ((unsigned)h3 << 16) };
      U2 pl = { (unsigned)l0 | ((unsigned)l1 << 16), (unsigned)l2 | ((unsigned)l3 << 16) };
      const int off = (r * 32 + a_row) * LDA + a_c4 * 4;
      *(U2 *)&sAh[off] = ph;
      *(U2 *)&sAl[off] = pl;
    }
    {
      unsigned hb[8], lb_[8];
      #pragma unroll
      for (int kk = 0; kk < 8; ++kk) {
        unsigned short h0, l0, h1, l1;
        split_bf16(bReg[2 * kk],     h0, l0);
        split_bf16(bReg[2 * kk + 1], h1, l1);
        hb[kk]  = (unsigned)h0 | ((unsigned)h1 << 16);
        lb_[kk] = (unsigned)l0 | ((unsigned)l1 << 16);
      }
      const int off = b_col * LDA + b_kh * 16;
      U4 *dh = (U4 *)&sBh[off];
      U4 *dl = (U4 *)&sBl[off];
      dh[0] = U4{hb[0], hb[1], hb[2], hb[3]};   dh[1] = U4{hb[4], hb[5], hb[6], hb[7]};
      dl[0] = U4{lb_[0], lb_[1], lb_[2], lb_[3]}; dl[1] = U4{lb_[4], lb_[5], lb_[6], lb_[7]};
    }
    __syncthreads();

    const int a_kbase = halfsel * 8;
    Frag fah[2], fal[2];
    #pragma unroll
    for (int mi = 0; mi < 2; ++mi) {
      const int r = (m0 + mi * 16 + lrow) * LDA;
      fah[mi].q[0] = *(const U4 *)&sAh[r + a_kbase];
      fah[mi].q[1] = *(const U4 *)&sAh[r + a_kbase + 16];
      fal[mi].q[0] = *(const U4 *)&sAl[r + a_kbase];
      fal[mi].q[1] = *(const U4 *)&sAl[r + a_kbase + 16];
    }
    const int b_kbase = halfsel * 8;
    #pragma unroll
    for (int ni = 0; ni < 4; ++ni) {
      const int c = (n0 + ni * 16 + lrow) * LDA;
      Frag fbh, fbl;
      fbh.q[0] = *(const U4 *)&sBh[c + b_kbase];
      fbh.q[1] = *(const U4 *)&sBh[c + b_kbase + 16];
      fbl.q[0] = *(const U4 *)&sBl[c + b_kbase];
      fbl.q[1] = *(const U4 *)&sBl[c + b_kbase + 16];
      #pragma unroll
      for (int mi = 0; mi < 2; ++mi) {
        acc[mi][ni] = __builtin_amdgcn_wmma_f32_16x16x32_bf16(
            false, fah[mi].v, false, fbh.v, (short)0, acc[mi][ni], false, false);
        acc[mi][ni] = __builtin_amdgcn_wmma_f32_16x16x32_bf16(
            false, fah[mi].v, false, fbl.v, (short)0, acc[mi][ni], false, false);
        acc[mi][ni] = __builtin_amdgcn_wmma_f32_16x16x32_bf16(
            false, fal[mi].v, false, fbh.v, (short)0, acc[mi][ni], false, false);
      }
    }
  }

  float midv[4], radv[4];
  #pragma unroll
  for (int ni = 0; ni < 4; ++ni) {
    const int c = i0 + n0 + ni * 16 + lrow;
    const float lo = lb0[c], hi = ub0[c];
    midv[ni] = 0.5f * (lo + hi);
    radv[ni] = 0.5f * (hi - lo);
  }
  #pragma unroll
  for (int mi = 0; mi < 2; ++mi) {
    #pragma unroll
    for (int v = 0; v < 8; ++v) {
      float lbp = 0.f, ubp = 0.f;
      #pragma unroll
      for (int ni = 0; ni < 4; ++ni) {
        const float m  = acc[mi][ni][v];
        const float t1 = m * midv[ni];
        const float t2 = fabsf(m) * radv[ni];
        lbp += t1 - t2;
        ubp += t1 + t2;
      }
      #pragma unroll
      for (int s = 8; s >= 1; s >>= 1) {
        lbp += __shfl_xor(lbp, s, 32);
        ubp += __shfl_xor(ubp, s, 32);
      }
      if (lrow == 0) {
        const int r = m0 + mi * 16 + v + halfsel * 8;
        atomicAdd(&sLB[r], lbp);
        atomicAdd(&sUB[r], ubp);
      }
    }
  }
  __syncthreads();
  if (t < BM) wsLB[(size_t)it * NI + j0 + t] = sLB[t];
  else        wsUB[(size_t)it * NI + j0 + (t - BM)] = sUB[t - BM];
}

// ---------------------------------------------------------------------------
// Kernel 5: sum chunk partials + bias, DeepPoly LeakyReLU relaxation, outputs.
// ---------------------------------------------------------------------------
__global__ void __launch_bounds__(256) finalize_kernel(
    const float *__restrict__ wsLB, const float *__restrict__ wsUB,
    const float *__restrict__ wsC,  const float *__restrict__ raw_alpha,
    float *__restrict__ out) {
  const int j = blockIdx.x * blockDim.x + threadIdx.x;
  float lb = 0.f, ub = 0.f;
  #pragma unroll 4
  for (int c = 0; c < KCHUNKS; ++c) {
    lb += wsLB[c * NI + j];
    ub += wsUB[c * NI + j];
  }
  const float cj = wsC[j];
  lb += cj;
  ub += cj;

  const float alpha = 1.0f / (1.0f + __expf(-raw_alpha[j]));
  const bool below = (ub <= 0.f);
  const bool above = (lb >= 0.f);
  const bool crossing = !(below || above);
  float s = (ub - lb * NS) / (ub - lb);
  if (s != s) s = 0.f;                                  // NaN -> 0 (reference semantics)
  const float uslope = crossing ? s : (above ? 1.f : (below ? NS : 1.f));
  const float uinter = crossing ? (1.f - s) * ub : 0.f;
  const float lbase  = (below && !above) ? NS : 1.f;
  const float v1 = crossing ? 1.f : lbase;
  const float v2 = crossing ? NS : lbase;
  const float lsl = alpha * v1 + (1.f - alpha) * v2;

  const size_t NN = (size_t)NI * NI;
  out[(size_t)j * NI + j]            = lsl;      // diag(lslope)
  out[NN + j]                        = 0.f;      // lintercept
  out[NN + NI + (size_t)j * NI + j]  = uslope;   // diag(uslope)
  out[NN + NI + NN + j]              = uinter;   // uintercept
}

// ---------------------------------------------------------------------------
extern "C" void kernel_launch(void *const *d_in, const int *in_sizes, int n_in,
                              void *d_out, int out_size, void *d_ws, size_t ws_size,
                              hipStream_t stream) {
  (void)in_sizes; (void)n_in;
  const float *lb0 = (const float *)d_in[0];
  const float *ub0 = (const float *)d_in[1];
  const float *W1  = (const float *)d_in[2];
  const float *b1  = (const float *)d_in[3];
  const float *W2  = (const float *)d_in[4];
  const float *b2  = (const float *)d_in[5];
  const float *raw_alpha = (const float *)d_in[6];
  float *out = (float *)d_out;

  float *ws   = (float *)d_ws;
  float *wsLB = ws;
  float *wsUB = ws + (size_t)KCHUNKS * NI;
  float *wsC  = ws + (size_t)2 * KCHUNKS * NI;
  const size_t partialFloats = (size_t)2 * KCHUNKS * NI + NI;
  const size_t planeElems = (size_t)NI * NI;
  unsigned short *Ah  = (unsigned short *)(ws + partialFloats);
  unsigned short *Al  = Ah + planeElems;
  unsigned short *BTh = Al + planeElems;
  unsigned short *BTl = BTh + planeElems;
  const size_t needed = partialFloats * sizeof(float) + 4 * planeElems * sizeof(unsigned short);
  const bool pre = (ws_size >= needed);

  const long n4 = (long)out_size / 4;
  fill_zero_kernel<<<dim3(2048), dim3(256), 0, stream>>>((float4 *)d_out, n4);
  bias_fold_kernel<<<dim3(NI / 8), dim3(256), 0, stream>>>(W2, b1, b2, wsC);

  if (pre) {
    split_w2_kernel<<<dim3((unsigned)(planeElems / 1024)), dim3(256), 0, stream>>>(W2, Ah, Al);
    split_w1t_kernel<<<dim3(NI / 64, NI / 64), dim3(256), 0, stream>>>(W1, BTh, BTl);
    gemm_bounds_async_kernel<<<dim3(NI / BM, NI / BN), dim3(256), 0, stream>>>(
        Ah, Al, BTh, BTl, lb0, ub0, wsLB, wsUB);
  } else {
    gemm_bounds_fused_kernel<<<dim3(NI / BM, NI / BN), dim3(256), 0, stream>>>(
        W2, W1, lb0, ub0, wsLB, wsUB);
  }
  finalize_kernel<<<dim3(NI / 256), dim3(256), 0, stream>>>(
      wsLB, wsUB, wsC, raw_alpha, out);
}
